// ConstrainedSUBLEQTransformer_52527450030361
// MI455X (gfx1250) — matrix operand order repaired and checked by reading, not compiled, not run on hardware
//
#include <hip/hip_runtime.h>

#define DEV __device__ __forceinline__

typedef __attribute__((ext_vector_type(16))) _Float16 v16h;
typedef __attribute__((ext_vector_type(8)))  float    v8f;

namespace {

constexpr int SEQ   = 33;
constexpr int DM    = 32;
constexpr int NH    = 8;
constexpr int FF    = 64;
constexpr int VOC   = 256;
constexpr int NL    = 4;
constexpr int NSEQ  = 4;              // sequences per workgroup
constexpr int TREAL = NSEQ * SEQ;     // 132 real tokens
constexpr int TPAD  = 144;            // padded to multiple of 16
constexpr int MT    = TPAD / 16;      // 9 M-tiles
constexpr int NTHREADS = 256;
constexpr int NWAVES   = 8;

// LDS layout (bytes), all 16B aligned
constexpr int OFF_H   = 0;                         // f32 [TPAD][32]  h (residual stream)
constexpr int OFF_HN  = OFF_H   + TPAD * DM * 4;   // f16 [TPAD][32]  layernorm output (A staging)
constexpr int OFF_QKV = OFF_HN  + TPAD * DM * 2;   // f32 [TPAD][96]  qkv
constexpr int OFF_AO  = OFF_QKV + TPAD * 96 * 4;   // f16 [TPAD][32]  attention out (A staging)
constexpr int OFF_FF  = OFF_AO  + TPAD * DM * 2;   // f16 [TPAD][64]  ffn hidden (A staging)
constexpr int SMEM_BYTES = OFF_FF + TPAD * FF * 2; // = 110592 bytes
static_assert(SMEM_BYTES <= 160 * 1024, "keep <=160KB for 2 WGs/WGP");

// A operand (16x32 f16): lane holds K=k0..k0+15 contiguous (k0 = 16*laneHi).
// p points at row base + k0 (+32*kstep); one contiguous 16-half run.
DEV v16h load_A16(const _Float16* p) {
  union { v16h v; uint4 q[2]; } a;
  a.q[0] = *(const uint4*)(p);
  a.q[1] = *(const uint4*)(p + 8);
  return a.v;
}

// B operand (32x16 f16): lane n = lane&15, lane-lo K=0..15, lane-hi K=16..31 (contiguous).
// p points at W + n*ldk + kstep*32 + 16*laneHi ; converts 16 f32 -> f16.
DEV v16h load_Bw(const float* p) {
  float4 f0 = *(const float4*)(p + 0);
  float4 f1 = *(const float4*)(p + 4);
  float4 f2 = *(const float4*)(p + 8);
  float4 f3 = *(const float4*)(p + 12);
  union { v16h v; _Float16 e[16]; } b;
  b.e[0]  = (_Float16)f0.x; b.e[1]  = (_Float16)f0.y; b.e[2]  = (_Float16)f0.z; b.e[3]  = (_Float16)f0.w;
  b.e[4]  = (_Float16)f1.x; b.e[5]  = (_Float16)f1.y; b.e[6]  = (_Float16)f1.z; b.e[7]  = (_Float16)f1.w;
  b.e[8]  = (_Float16)f2.x; b.e[9]  = (_Float16)f2.y; b.e[10] = (_Float16)f2.z; b.e[11] = (_Float16)f2.w;
  b.e[12] = (_Float16)f3.x; b.e[13] = (_Float16)f3.y; b.e[14] = (_Float16)f3.z; b.e[15] = (_Float16)f3.w;
  return b.v;
}

DEV v8f wmma16(v16h a, v16h b, v8f c) {
  // (neg_a, A, neg_b, B, c_mod, C, reuse_a, reuse_b)
  return __builtin_amdgcn_wmma_f32_16x16x32_f16(false, a, false, b, (short)0, c, false, false);
}

// GEMM: D[M=TPAD][N=16*n_tiles] = A[M][32*KSTEPS] * W^T, W row-major [N][32*KSTEPS] (ldk).
// Wave-level tiles: tasks = n_tiles * MSPLIT, wave-uniform loop => EXEC all-1s at WMMA.
template <int KSTEPS, int MSPLIT, typename Epi>
DEV void gemm_tiles(const _Float16* A, int lda, const float* W, int ldk, int n_tiles, Epi epi) {
  const int tid    = threadIdx.x;
  const int wave   = tid >> 5;
  const int lane   = tid & 31;
  const int laneHi = lane >> 4;
  const int nloc   = lane & 15;
  const int mchunk = (MT + MSPLIT - 1) / MSPLIT;
  const int ntasks = n_tiles * MSPLIT;
  for (int task = wave; task < ntasks; task += NWAVES) {
    const int nt = task % n_tiles;
    const int mc = task / n_tiles;
    const int m0 = mc * mchunk;
    const int m1 = (m0 + mchunk < MT) ? (m0 + mchunk) : MT;
    v16h B[KSTEPS];
#pragma unroll
    for (int ks = 0; ks < KSTEPS; ++ks)
      B[ks] = load_Bw(W + (nt * 16 + nloc) * ldk + ks * 32 + laneHi * 16);
    for (int mt = m0; mt < m1; ++mt) {
      v8f c = {};
#pragma unroll
      for (int ks = 0; ks < KSTEPS; ++ks) {
        v16h a = load_A16(A + (mt * 16 + nloc) * lda + ks * 32 + laneHi * 16);
        c = wmma16(a, B[ks], c);
      }
#pragma unroll
      for (int v = 0; v < 8; ++v)
        epi(mt * 16 + v + laneHi * 8, nt * 16 + nloc, c[v]);
    }
  }
}

DEV void layernorm(const float* h, _Float16* hn, const float* w, const float* b) {
  for (int t = threadIdx.x; t < TPAD; t += NTHREADS) {
    const float* row = h + t * DM;
    float m = 0.f;
#pragma unroll
    for (int d = 0; d < DM; ++d) m += row[d];
    m *= (1.f / DM);
    float var = 0.f;
#pragma unroll
    for (int d = 0; d < DM; ++d) { float df = row[d] - m; var += df * df; }
    var *= (1.f / DM);
    float r = rsqrtf(var + 1e-5f);
#pragma unroll
    for (int d = 0; d < DM; ++d)
      hn[t * DM + d] = (_Float16)((row[d] - m) * r * w[d] + b[d]);
  }
}

} // namespace

__global__ __launch_bounds__(NTHREADS, 1)
void tinybert_fused(const int* __restrict__ x,
                    const float* token_emb, const float* pos_emb, const float* type_emb,
                    const float* ln1_w, const float* ln1_b,
                    const float* qkv_w, const float* qkv_b,
                    const float* out_w, const float* out_b,
                    const float* ln2_w, const float* ln2_b,
                    const float* w1, const float* b1,
                    const float* w2, const float* b2,
                    const float* lnf_w, const float* lnf_b,
                    const float* head_w, const float* head_b,
                    float* __restrict__ out) {
  extern __shared__ char smem[];
  float*    h   = (float*)(smem + OFF_H);
  _Float16* hn  = (_Float16*)(smem + OFF_HN);
  float*    qkv = (float*)(smem + OFF_QKV);
  _Float16* ao  = (_Float16*)(smem + OFF_AO);
  _Float16* ff  = (_Float16*)(smem + OFF_FF);
  const int wg  = blockIdx.x;
  const int tid = threadIdx.x;

  // ---- embeddings: h[t][d] = tok + pos + type, pad rows zeroed ----
  for (int idx = tid; idx < TPAD * DM; idx += NTHREADS) {
    int t = idx >> 5, d = idx & 31;
    float val = 0.f;
    if (t < TREAL) {
      int s   = t % SEQ;
      int tok = x[wg * NSEQ * SEQ + t];
      val = token_emb[tok * DM + d] + pos_emb[s * DM + d] + type_emb[(s > 0 ? 1 : 0) * DM + d];
    }
    h[idx] = val;
  }
  __syncthreads();

  for (int l = 0; l < NL; ++l) {
    const float* qw  = qkv_w + l * 96 * DM;  const float* qb  = qkv_b + l * 96;
    const float* ow  = out_w + l * DM * DM;  const float* ob  = out_b + l * DM;
    const float* w1l = w1    + l * FF * DM;  const float* b1l = b1    + l * FF;
    const float* w2l = w2    + l * DM * FF;  const float* b2l = b2    + l * DM;

    // ---- pre-LN + QKV projection: [TPAD,32] x [32,96] ----
    layernorm(h, hn, ln1_w + l * DM, ln1_b + l * DM);
    __syncthreads();
    gemm_tiles<1, 2>(hn, DM, qw, DM, 96 / 16,
      [&](int row, int col, float v) { qkv[row * 96 + col] = v + qb[col]; });
    __syncthreads();

    // ---- attention: d_head=4, S=33; one (seq,head,query) row per thread ----
    for (int r = tid; r < NSEQ * NH * SEQ; r += NTHREADS) {
      int seq  = r / (NH * SEQ);
      int rem  = r % (NH * SEQ);
      int head = rem / SEQ;
      int i    = rem % SEQ;
      int ti   = seq * SEQ + i;
      float4 q = *(const float4*)&qkv[ti * 96 + head * 4];
      float sc[SEQ];
#pragma unroll
      for (int j = 0; j < SEQ; ++j) {
        float4 k = *(const float4*)&qkv[(seq * SEQ + j) * 96 + 32 + head * 4];
        sc[j] = 0.5f * (q.x * k.x + q.y * k.y + q.z * k.z + q.w * k.w);
      }
      float mx = sc[0];
#pragma unroll
      for (int j = 1; j < SEQ; ++j) mx = fmaxf(mx, sc[j]);
      float sum = 0.f;
#pragma unroll
      for (int j = 0; j < SEQ; ++j) { sc[j] = __expf(sc[j] - mx); sum += sc[j]; }
      float inv = 1.f / sum;
      float o0 = 0.f, o1 = 0.f, o2 = 0.f, o3 = 0.f;
#pragma unroll
      for (int j = 0; j < SEQ; ++j) {
        float4 vv = *(const float4*)&qkv[(seq * SEQ + j) * 96 + 64 + head * 4];
        o0 += sc[j] * vv.x; o1 += sc[j] * vv.y; o2 += sc[j] * vv.z; o3 += sc[j] * vv.w;
      }
      ao[ti * DM + head * 4 + 0] = (_Float16)(o0 * inv);
      ao[ti * DM + head * 4 + 1] = (_Float16)(o1 * inv);
      ao[ti * DM + head * 4 + 2] = (_Float16)(o2 * inv);
      ao[ti * DM + head * 4 + 3] = (_Float16)(o3 * inv);
    }
    __syncthreads();

    // ---- output projection + residual: [TPAD,32] x [32,32] ----
    gemm_tiles<1, 4>(ao, DM, ow, DM, DM / 16,
      [&](int row, int col, float v) { h[row * DM + col] += v + ob[col]; });
    __syncthreads();

    // ---- pre-LN + FFN up (relu): [TPAD,32] x [32,64] ----
    layernorm(h, hn, ln2_w + l * DM, ln2_b + l * DM);
    __syncthreads();
    gemm_tiles<1, 2>(hn, DM, w1l, DM, FF / 16,
      [&](int row, int col, float v) {
        float t2 = v + b1l[col];
        ff[row * FF + col] = (_Float16)(t2 > 0.f ? t2 : 0.f);
      });
    __syncthreads();

    // ---- FFN down + residual: [TPAD,64] x [64,32], two K-steps ----
    gemm_tiles<2, 4>(ff, FF, w2l, FF, DM / 16,
      [&](int row, int col, float v) { h[row * DM + col] += v + b2l[col]; });
    __syncthreads();
  }

  // ---- final LN + vocab head: [TPAD,32] x [32,256] -> HBM (single pass) ----
  layernorm(h, hn, lnf_w, lnf_b);
  __syncthreads();
  gemm_tiles<1, 1>(hn, DM, head_w, DM, VOC / 16,
    [&](int row, int col, float v) {
      if (row < TREAL)
        out[(size_t)(wg * TREAL + row) * VOC + col] = v + head_b[col];
    });
}

extern "C" void kernel_launch(void* const* d_in, const int* in_sizes, int n_in,
                              void* d_out, int out_size, void* d_ws, size_t ws_size,
                              hipStream_t stream) {
  (void)n_in; (void)out_size; (void)d_ws; (void)ws_size;
  const int*   x         = (const int*)d_in[0];
  const float* token_emb = (const float*)d_in[1];
  const float* pos_emb   = (const float*)d_in[2];
  const float* type_emb  = (const float*)d_in[3];
  const float* ln1_w     = (const float*)d_in[4];
  const float* ln1_b     = (const float*)d_in[5];
  const float* qkv_w     = (const float*)d_in[6];
  const float* qkv_b     = (const float*)d_in[7];
  const float* out_w     = (const float*)d_in[8];
  const float* out_b     = (const float*)d_in[9];
  const float* ln2_w     = (const float*)d_in[10];
  const float* ln2_b     = (const float*)d_in[11];
  const float* w1        = (const float*)d_in[12];
  const float* b1        = (const float*)d_in[13];
  const float* w2        = (const float*)d_in[14];
  const float* b2        = (const float*)d_in[15];
  const float* lnf_w     = (const float*)d_in[16];
  const float* lnf_b     = (const float*)d_in[17];
  const float* head_w    = (const float*)d_in[18];
  const float* head_b    = (const float*)d_in[19];

  const int B      = in_sizes[0] / SEQ;   // 8192
  const int blocks = B / NSEQ;            // 2048 workgroups x 4 sequences

  tinybert_fused<<<blocks, NTHREADS, SMEM_BYTES, stream>>>(
      x, token_emb, pos_emb, type_emb,
      ln1_w, ln1_b, qkv_w, qkv_b, out_w, out_b,
      ln2_w, ln2_b, w1, b1, w2, b2,
      lnf_w, lnf_b, head_w, head_b,
      (float*)d_out);
}
